// LearnableNodeSelector_85779086836037
// MI455X (gfx1250) — matrix operand order repaired and checked by reading, not compiled, not run on hardware
//
#include <hip/hip_runtime.h>
#include <hip/hip_bf16.h>
#include <math.h>

typedef __attribute__((ext_vector_type(16))) _Float16 v16h;
typedef __attribute__((ext_vector_type(8)))  float    v8f;

#define BSZ   512
#define NN    4096
#define TOPK  36
#define TPW   8                       // tiles per wave
#define NEG_INF (-__builtin_inff())

// ---------------------------------------------------------------- init
__global__ void lns_init_kernel(unsigned* maxbits) {
    if (threadIdx.x == 0) maxbits[0] = 0u;   // bits of +0.0f
}

// ---------------------------------------------------------------- global abs-max of offsets
__global__ void lns_absmax_kernel(const float4* __restrict__ x, int n4,
                                  unsigned* __restrict__ out) {
    __shared__ float red[256];
    int t = threadIdx.x;
    float mm = 0.0f;
    for (long long i = (long long)blockIdx.x * 256 + t; i < n4;
         i += (long long)gridDim.x * 256) {
        float4 v = x[i];
        mm = fmaxf(mm, fmaxf(fmaxf(fabsf(v.x), fabsf(v.y)),
                             fmaxf(fabsf(v.z), fabsf(v.w))));
    }
    red[t] = mm;
    __syncthreads();
    for (int s = 128; s > 0; s >>= 1) {
        if (t < s) red[t] = fmaxf(red[t], red[t + s]);
        __syncthreads();
    }
    if (t == 0) atomicMax(out, __float_as_uint(red[0]));
}

// A-fragment K index for 16-bit 16x32 A matrix (ISA 7.12.2)
__device__ __forceinline__ int amapK(int e, int hi) {
    int v = e >> 1, s = e & 1;
    return ((v & 4) ? 16 : 0) + (hi ? 8 : 0) + ((v & 3) << 1) + s;
}

// ---------------------------------------------------------------- main scoring kernel
__global__ void __launch_bounds__(128)
lns_score_kernel(const float* __restrict__ feat,   // [B*N,4]
                 const float* __restrict__ offs,   // [B*N,3]
                 const int*   __restrict__ mask,   // [B*N]
                 const float* __restrict__ Wpe1,   // [3,32]
                 const float* __restrict__ bpe1,   // [32]
                 const float* __restrict__ Wpe2,   // [32,64]
                 const float* __restrict__ bpe2,   // [64]
                 const float* __restrict__ Wgate,  // [68,64]
                 const float* __restrict__ bgate,  // [64]
                 const float* __restrict__ Whead,  // [64]
                 const float* __restrict__ bhead,  // [1]
                 const unsigned* __restrict__ maxbits,
                 float* __restrict__ scores)       // [B*N]
{
    // 16 pre-swizzled B fragments (frag 0..3: Wpe2 col-groups; 4..7/8..11/12..15:
    // Wgate K-chunks 0/1/2 per col-group). Layout: [frag][lane][16 halfs] so a
    // fragment load is one v16h (2x ds_load_b128) per lane.
    __shared__ __align__(32) _Float16 sB[16 * 32 * 16];        // 16 KB
    // per-wave A-fragment staging for gate chunks 0/1 (f16, A layout)
    __shared__ __align__(32) _Float16 sA[4 * 2 * 32 * 16];     // 8 KB
    // packed pe1 weights: (W1[0][k], W1[1][k], W1[2][k], b1[k])
    __shared__ float4 sW1[32];

    const int t = threadIdx.x;
    if (t < 32) sW1[t] = make_float4(Wpe1[t], Wpe1[32 + t], Wpe1[64 + t], bpe1[t]);
    for (int idx = t; idx < 16 * 32 * 16; idx += 128) {
        int f    = idx >> 9;
        int rem  = idx & 511;
        int ln   = rem >> 4;
        int e    = rem & 15;
        int kloc = ((e >> 3) << 4) + ((ln >> 4) << 3) + (e & 7);  // B layout: K = 16*(e/8) + 8*hi + e%8
        int nloc = ln & 15;                   // N = lane % 16
        float v;
        if (f < 4) {
            v = Wpe2[kloc * 64 + f * 16 + nloc];
        } else {
            int c  = (f - 4) >> 2;            // K-chunk 0..2
            int gc = (f - 4) & 3;             // col-group
            int row = c * 32 + kloc;          // combined input row
            v = (row < 68) ? Wgate[row * 64 + gc * 16 + nloc] : 0.0f;
        }
        sB[idx] = (_Float16)v;
    }
    __syncthreads();

    const v16h* Bf = (const v16h*)sB;         // fragment f, lane l -> Bf[f*32+l]
    const v16h* Af = (const v16h*)sA;

    const int wave = t >> 5, lane = t & 31;
    const int m  = lane & 15;
    const int hi = lane >> 4;

    // tile-invariant scalars (cached global loads)
    const float maxr = fmaxf(__uint_as_float(maxbits[0]), 1.0f);
    const float invr = 1.0f / maxr;
    const float bh0  = bhead[0];
    float cb2[4], cbg[4], cwh[4];
#pragma unroll
    for (int gc = 0; gc < 4; ++gc) {
        cb2[gc] = bpe2[gc * 16 + m];
        cbg[gc] = bgate[gc * 16 + m];
        cwh[gc] = Whead[gc * 16 + m];
    }

    const int baseTile = (blockIdx.x * 4 + wave) * TPW;

#pragma unroll 1
    for (int it = 0; it < TPW; ++it) {
        const int g0   = (baseTile + it) * 16;
        const int node = g0 + m;

        const float ox = offs[node * 3 + 0] * invr;
        const float oy = offs[node * 3 + 1] * invr;
        const float oz = offs[node * 3 + 2] * invr;

        // ---- h = gelu(norm_off @ Wpe1 + b1), built directly in A layout
        v16h aPE;
#pragma unroll
        for (int e = 0; e < 16; ++e) {
            int k = amapK(e, hi);
            float4 w = sW1[k];
            float x = ox * w.x + oy * w.y + oz * w.z + w.w;
            float g = 0.5f * x * (1.0f + erff(x * 0.70710678118654752f));
            aPE[e] = (_Float16)g;
        }

        // ---- pos_enc = h @ Wpe2 + b2 : 4 WMMAs
        v8f posD[4];
#pragma unroll
        for (int gc = 0; gc < 4; ++gc) {
            v16h bF = Bf[gc * 32 + lane];
            v8f c;
#pragma unroll
            for (int r = 0; r < 8; ++r) c[r] = cb2[gc];
            posD[gc] = __builtin_amdgcn_wmma_f32_16x16x32_f16(
                false, aPE, false, bF, (short)0, c, false, false);
        }

        // ---- D layout -> A layout (f16) staged in per-wave LDS.
        // col = gc*16+m  =>  chunk = gc>>1, e = (gc&1)*8 | (m&7), hi' = m>>3,
        // dest lane = (r + 8*hi) + 16*hi'  (stride 16 halfs over r)
#pragma unroll
        for (int gc = 0; gc < 4; ++gc) {
            int chunk   = gc >> 1;
            int e       = ((gc & 1) << 3) | (m & 7);
            int laneDst = (hi << 3) + ((m >> 3) << 4);
            _Float16* dst = sA + (((wave * 2 + chunk) * 32 + laneDst) << 4) + e;
#pragma unroll
            for (int r = 0; r < 8; ++r) dst[r * 16] = (_Float16)posD[gc][r];
        }
        asm volatile("s_wait_dscnt 0x0" ::: "memory");  // wave-local LDS RAW

        v16h aG0 = Af[(wave * 2 + 0) * 32 + lane];
        v16h aG1 = Af[(wave * 2 + 1) * 32 + lane];

        // feature chunk: rows 64..67 of combined input; only lanes 0-15, e=0..3
        const float4 f4 = ((const float4*)feat)[node];
        v16h aG2 = {};
        aG2[0] = (_Float16)(hi ? 0.0f : f4.x);
        aG2[1] = (_Float16)(hi ? 0.0f : f4.y);
        aG2[2] = (_Float16)(hi ? 0.0f : f4.z);
        aG2[3] = (_Float16)(hi ? 0.0f : f4.w);

        // ---- gate = sigmoid([pos|feat] @ Wgate + bg); fused = gate*pos; head dot
        float p[8];
#pragma unroll
        for (int r = 0; r < 8; ++r) p[r] = 0.0f;

#pragma unroll
        for (int gc = 0; gc < 4; ++gc) {
            v8f acc;
#pragma unroll
            for (int r = 0; r < 8; ++r) acc[r] = cbg[gc];
            acc = __builtin_amdgcn_wmma_f32_16x16x32_f16(
                false, aG0, false, Bf[(4 + gc) * 32 + lane],  (short)0, acc, false, false);
            acc = __builtin_amdgcn_wmma_f32_16x16x32_f16(
                false, aG1, false, Bf[(8 + gc) * 32 + lane],  (short)0, acc, false, false);
            acc = __builtin_amdgcn_wmma_f32_16x16x32_f16(
                false, aG2, false, Bf[(12 + gc) * 32 + lane], (short)0, acc, false, false);
            const float wh = cwh[gc];
#pragma unroll
            for (int r = 0; r < 8; ++r) {
                float gate  = 1.0f / (1.0f + __expf(-acc[r]));
                p[r] += gate * posD[gc][r] * wh;
            }
        }

        // ---- reduce head partials across the 16 columns of each half-wave
#pragma unroll
        for (int off = 8; off >= 1; off >>= 1)
#pragma unroll
            for (int r = 0; r < 8; ++r)
                p[r] += __shfl_xor(p[r], off, 16);

        if (m == 0) {
#pragma unroll
            for (int r = 0; r < 8; ++r) {
                int nd  = g0 + r + 8 * hi;
                float s = p[r] + bh0;
                scores[nd] = mask[nd] ? s : NEG_INF;
            }
        }
    }
}

// ---------------------------------------------------------------- softmax + top-36 per row
__global__ void __launch_bounds__(256)
lns_softmax_topk_kernel(const float* __restrict__ scores,
                        int*   __restrict__ oIdx,
                        float* __restrict__ oVal,
                        float* __restrict__ oAttn)
{
    __shared__ float ls[NN];
    __shared__ float rv[256];
    __shared__ int   ri[256];

    const int b = blockIdx.x, t = threadIdx.x;
    const float* row = scores + (long long)b * NN;
    for (int i = t; i < NN; i += 256) ls[i] = row[i];
    __syncthreads();

    // row max
    float mx = NEG_INF;
    for (int i = t; i < NN; i += 256) mx = fmaxf(mx, ls[i]);
    rv[t] = mx; __syncthreads();
    for (int s = 128; s > 0; s >>= 1) {
        if (t < s) rv[t] = fmaxf(rv[t], rv[t + s]);
        __syncthreads();
    }
    mx = rv[0];
    __syncthreads();

    // sum of exp
    float sm = 0.0f;
    for (int i = t; i < NN; i += 256) {
        float v = ls[i];
        sm += (v == NEG_INF) ? 0.0f : expf(v - mx);
    }
    rv[t] = sm; __syncthreads();
    for (int s = 128; s > 0; s >>= 1) {
        if (t < s) rv[t] += rv[t + s];
        __syncthreads();
    }
    const float tot = rv[0];
    const float inv = (tot > 0.0f) ? 1.0f / tot : 0.0f;
    __syncthreads();

    float* aw = oAttn + (long long)b * NN;
    for (int i = t; i < NN; i += 256) {
        float v = ls[i];
        aw[i] = (v == NEG_INF) ? 0.0f : expf(v - mx) * inv;
    }

    // iterative top-36 (ties -> smaller index, matching jax top_k)
    for (int j = 0; j < TOPK; ++j) {
        float bv = NEG_INF;
        int   bi = NN;
        for (int i = t; i < NN; i += 256) {
            if (ls[i] > bv) { bv = ls[i]; bi = i; }
        }
        rv[t] = bv; ri[t] = bi;
        __syncthreads();
        for (int s = 128; s > 0; s >>= 1) {
            if (t < s) {
                if (rv[t + s] > rv[t] ||
                    (rv[t + s] == rv[t] && ri[t + s] < ri[t])) {
                    rv[t] = rv[t + s];
                    ri[t] = ri[t + s];
                }
            }
            __syncthreads();
        }
        if (t == 0) {
            oIdx[b * TOPK + j] = ri[0];
            oVal[b * TOPK + j] = rv[0];
            if (ri[0] < NN) ls[ri[0]] = NEG_INF;
        }
        __syncthreads();
    }
}

// ---------------------------------------------------------------- launch
extern "C" void kernel_launch(void* const* d_in, const int* in_sizes, int n_in,
                              void* d_out, int out_size, void* d_ws, size_t ws_size,
                              hipStream_t stream) {
    const float* feat  = (const float*)d_in[0];
    const float* offs  = (const float*)d_in[1];
    const int*   mask  = (const int*)d_in[2];
    const float* Wpe1  = (const float*)d_in[3];
    const float* bpe1  = (const float*)d_in[4];
    const float* Wpe2  = (const float*)d_in[5];
    const float* bpe2  = (const float*)d_in[6];
    const float* Wgate = (const float*)d_in[7];
    const float* bgate = (const float*)d_in[8];
    const float* Whead = (const float*)d_in[9];
    const float* bhead = (const float*)d_in[10];

    unsigned* maxbits = (unsigned*)d_ws;
    float*    scores  = (float*)d_ws + 64;          // +256 B past the scalar

    int*   oIdx  = (int*)d_out;
    float* oVal  = (float*)d_out + BSZ * TOPK;
    float* oAttn = (float*)d_out + 2 * BSZ * TOPK;

    lns_init_kernel<<<1, 1, 0, stream>>>(maxbits);
    lns_absmax_kernel<<<2048, 256, 0, stream>>>((const float4*)offs,
                                                (BSZ * NN * 3) / 4, maxbits);

    const int tiles  = (BSZ * NN) / 16;             // 131072
    const int blocks = tiles / (4 * TPW);           // 4 waves/block, TPW tiles/wave
    lns_score_kernel<<<blocks, 128, 0, stream>>>(
        feat, offs, mask, Wpe1, bpe1, Wpe2, bpe2, Wgate, bgate, Whead, bhead,
        maxbits, scores);

    lns_softmax_topk_kernel<<<BSZ, 256, 0, stream>>>(scores, oIdx, oVal, oAttn);
}
